// DiffusionModel_37022618092046
// MI455X (gfx1250) — matrix operand order repaired and checked by reading, not compiled, not run on hardware
//
#include <hip/hip_runtime.h>
#include <math.h>

// ---- model constants ----
constexpr int Bn = 4, Nn = 24, NDn = 13, EDn = 12;
constexpr int Hn = 256, NHn = 8, HDn = 32, Ln = 64;
constexpr float SCALEC = 0.0625f;           // 1/sqrt(256)
constexpr int ROWS_N = Bn * Nn;             // 96
constexpr int ROWS_E = Bn * Nn * Nn;        // 2304

typedef __attribute__((ext_vector_type(16))) _Float16 v16h;
typedef __attribute__((ext_vector_type(8)))  float    v8f;

// ---- wave32 reductions ----
__device__ inline float wsum(float v) {
    #pragma unroll
    for (int o = 16; o; o >>= 1) v += __shfl_xor(v, o, 32);
    return v;
}
__device__ inline float wmax(float v) {
    #pragma unroll
    for (int o = 16; o; o >>= 1) v = fmaxf(v, __shfl_xor(v, o, 32));
    return v;
}

// load one 16-bit-A-layout fragment row from f32 memory
__device__ inline v16h load_a_frag(const float* __restrict__ rowp) {
    const float4* ap = reinterpret_cast<const float4*>(rowp);
    const float4 a0 = ap[0];   // k+0..3
    const float4 a1 = ap[1];   // k+4..7
    const float4 a2 = ap[4];   // k+16..19
    const float4 a3 = ap[5];   // k+20..23
    v16h af;
    af[0]  = (_Float16)a0.x; af[1]  = (_Float16)a0.y;
    af[2]  = (_Float16)a0.z; af[3]  = (_Float16)a0.w;
    af[4]  = (_Float16)a1.x; af[5]  = (_Float16)a1.y;
    af[6]  = (_Float16)a1.z; af[7]  = (_Float16)a1.w;
    af[8]  = (_Float16)a2.x; af[9]  = (_Float16)a2.y;
    af[10] = (_Float16)a2.z; af[11] = (_Float16)a2.w;
    af[12] = (_Float16)a3.x; af[13] = (_Float16)a3.y;
    af[14] = (_Float16)a3.z; af[15] = (_Float16)a3.w;
    return af;
}

// =====================================================================
// WMMA GEMM: out = epilogue(A[MxK] @ W[KxN] + bias[N])
// MODE bit0: relu(acc+bias); bit1: += res[MxN] (applied after relu)
// grid = (M/32, N/128), block = 128: 4 waves, each wave owns a 32x32
// macro-tile = 2 M-tiles x 2 N-tiles (4 WMMA per k-step, A and B
// fragments each reused twice -> half the loads per WMMA).
// =====================================================================
template<int MODE>
__global__ __launch_bounds__(128)
void gemm_wmma_kernel(const float* __restrict__ A, const float* __restrict__ W,
                      const float* __restrict__ bias, const float* __restrict__ res,
                      float* __restrict__ out, int M, int N, int K)
{
    const int lane = threadIdx.x & 31;
    const int wave = threadIdx.x >> 5;
    const int m0 = blockIdx.x * 32;
    const int n0 = (blockIdx.y * 4 + wave) * 32;
    const int hl = lane & 15;          // row within A tile / col within B tile
    const int hi = lane >> 4;          // half-wave select
    const int arow0 = m0 + hl;
    const int arow1 = arow0 + 16;
    const int bcol0 = n0 + hl;
    const int bcol1 = bcol0 + 16;

    v8f acc00 = {}, acc01 = {}, acc10 = {}, acc11 = {};
    #pragma unroll 2
    for (int kb = 0; kb < K; kb += 32) {
        // A fragments: 16-bit A 16x32 layout (ISA 7.12.2)
        // lanes 0-15: elems 0..7 = K kb+0..7, elems 8..15 = K kb+16..23
        // lanes16-31: elems 0..7 = K kb+8..15, elems 8..15 = K kb+24..31
        const int ka = kb + hi * 8;
        const v16h af0 = load_a_frag(A + (size_t)arow0 * K + ka);
        const v16h af1 = load_a_frag(A + (size_t)arow1 * K + ka);

        // B fragments: same K order as A (lanes 0-15: kb+0..7,16..23; lanes 16-31: kb+8..15,24..31)
        const float* wp = W + (size_t)(kb + hi * 8) * N;
        v16h bf0, bf1;
        #pragma unroll
        for (int i = 0; i < 16; ++i) {
            const float* wr = wp + (size_t)(i + (i >> 3) * 8) * N;
            bf0[i] = (_Float16)wr[bcol0];
            bf1[i] = (_Float16)wr[bcol1];
        }
        acc00 = __builtin_amdgcn_wmma_f32_16x16x32_f16(
                    false, af0, false, bf0, (short)0, acc00, false, false);
        acc01 = __builtin_amdgcn_wmma_f32_16x16x32_f16(
                    false, af0, false, bf1, (short)0, acc01, false, false);
        acc10 = __builtin_amdgcn_wmma_f32_16x16x32_f16(
                    false, af1, false, bf0, (short)0, acc10, false, false);
        acc11 = __builtin_amdgcn_wmma_f32_16x16x32_f16(
                    false, af1, false, bf1, (short)0, acc11, false, false);
    }

    // C/D layout: lanes 0-15 col=tile_n+lane rows tile_m+r; lanes 16-31 rows tile_m+8+r
    const int rb0 = m0 + hi * 8;        // rows of M-tile 0
    const int rb1 = rb0 + 16;           // rows of M-tile 1
    const float bv0 = bias[bcol0];
    const float bv1 = bias[bcol1];
    #pragma unroll
    for (int r = 0; r < 8; ++r) {
        const size_t ro0 = (size_t)(rb0 + r) * N;
        const size_t ro1 = (size_t)(rb1 + r) * N;
        float v00 = acc00[r] + bv0;
        float v01 = acc01[r] + bv1;
        float v10 = acc10[r] + bv0;
        float v11 = acc11[r] + bv1;
        if constexpr (MODE & 1) {
            v00 = fmaxf(v00, 0.f); v01 = fmaxf(v01, 0.f);
            v10 = fmaxf(v10, 0.f); v11 = fmaxf(v11, 0.f);
        }
        if constexpr (MODE & 2) {
            v00 += res[ro0 + bcol0]; v01 += res[ro0 + bcol1];
            v10 += res[ro1 + bcol0]; v11 += res[ro1 + bcol1];
        }
        out[ro0 + bcol0] = v00;
        out[ro0 + bcol1] = v01;
        out[ro1 + bcol0] = v10;
        out[ro1 + bcol1] = v11;
    }
}

// =====================================================================
// Attention: one wave per (b, i, head); lane = head-dim d (HD==32==wave)
// logits[j] = SCALE * q.k + sum_d e ; softmax over j (N=24); wv = attn @ v
// =====================================================================
__global__ __launch_bounds__(128)
void attn_kernel(const float* __restrict__ qkv, const float* __restrict__ e,
                 float* __restrict__ wv)
{
    const int gw = (blockIdx.x * blockDim.x + threadIdx.x) >> 5;
    const int lane = threadIdx.x & 31;
    const int h = gw % NHn;
    const int i = (gw / NHn) % Nn;
    const int b = gw / (NHn * Nn);
    const int hd = h * HDn + lane;

    const float qd = qkv[(size_t)(b * Nn + i) * 768 + hd];
    float logit = 0.f;
    for (int j = 0; j < Nn; ++j) {
        float kd = qkv[(size_t)(b * Nn + j) * 768 + 256 + hd];
        float ed = e[((size_t)(b * Nn + i) * Nn + j) * Hn + hd];
        float t = wsum(qd * kd * SCALEC + ed);
        if (lane == j) logit = t;
    }
    float ml = (lane < Nn) ? logit : -3.0e38f;
    float mx = wmax(ml);
    float ex = (lane < Nn) ? __expf(logit - mx) : 0.f;
    float s = wsum(ex);
    float a = ex / s;

    float accv = 0.f;
    for (int j = 0; j < Nn; ++j) {
        float aj = __shfl(a, j, 32);
        accv += aj * qkv[(size_t)(b * Nn + j) * 768 + 512 + hd];
    }
    wv[(size_t)(b * Nn + i) * Hn + hd] = accv;
}

// ---- LayerNorm over last dim (256), one row per block ----
__global__ __launch_bounds__(256)
void ln_kernel(const float* __restrict__ X, const float* __restrict__ g,
               const float* __restrict__ bt, float* __restrict__ out)
{
    __shared__ float sd[256];
    const int row = blockIdx.x, t = threadIdx.x;
    float x = X[(size_t)row * Hn + t];
    sd[t] = x; __syncthreads();
    for (int s = 128; s; s >>= 1) { if (t < s) sd[t] += sd[t + s]; __syncthreads(); }
    float mean = sd[0] * (1.f / Hn); __syncthreads();
    float dx = x - mean;
    sd[t] = dx * dx; __syncthreads();
    for (int s = 128; s; s >>= 1) { if (t < s) sd[t] += sd[t + s]; __syncthreads(); }
    float var = sd[0] * (1.f / Hn);
    out[(size_t)row * Hn + t] = dx * rsqrtf(var + 1e-5f) * g[t] + bt[t];
}

// ---- cond MLP: one-hot row select + 2-layer MLP, one block per batch ----
__global__ __launch_bounds__(256)
void cond_kernel(const int* __restrict__ conds, const float* __restrict__ w1,
                 const float* __restrict__ b1, const float* __restrict__ w2,
                 const float* __restrict__ b2, float* __restrict__ c)
{
    __shared__ float h1[256];
    const int b = blockIdx.x, t = threadIdx.x;
    const int tok = conds[b];
    h1[t] = fmaxf(w1[(size_t)tok * Hn + t] + b1[t], 0.f);
    __syncthreads();
    float acc = b2[t];
    for (int k = 0; k < Hn; ++k) acc += h1[k] * w2[(size_t)k * Hn + t];
    c[(size_t)b * Hn + t] = fmaxf(acc, 0.f);
}

// ---- node input MLP (K=13 then K=256), one block per row ----
__global__ __launch_bounds__(256)
void nodein_kernel(const float* __restrict__ nodes, const float* __restrict__ w1,
                   const float* __restrict__ b1, const float* __restrict__ w2,
                   const float* __restrict__ b2, float* __restrict__ n)
{
    __shared__ float h1[256];
    const int row = blockIdx.x, t = threadIdx.x;
    float acc = b1[t];
    for (int k = 0; k < NDn; ++k) acc += nodes[row * NDn + k] * w1[k * Hn + t];
    h1[t] = fmaxf(acc, 0.f);
    __syncthreads();
    acc = b2[t];
    for (int k = 0; k < Hn; ++k) acc += h1[k] * w2[k * Hn + t];
    n[(size_t)row * Hn + t] = fmaxf(acc, 0.f);
}

// ---- edge input stage 1: relu(edges @ w1 + b1), K=12 elementwise ----
__global__ __launch_bounds__(256)
void edgein1_kernel(const float* __restrict__ edges, const float* __restrict__ w1,
                    const float* __restrict__ b1, float* __restrict__ E1)
{
    const size_t idx = (size_t)blockIdx.x * 256 + threadIdx.x;  // ROWS_E*H exact
    const int col = idx & 255;
    const size_t row = idx >> 8;
    float acc = b1[col];
    const float* er = edges + row * EDn;
    for (int k = 0; k < EDn; ++k) acc += er[k] * w1[k * Hn + col];
    E1[idx] = fmaxf(acc, 0.f);
}

// ---- nc = n + c (broadcast over batch) ----
__global__ __launch_bounds__(256)
void addc_kernel(const float* __restrict__ n, const float* __restrict__ c,
                 float* __restrict__ nc)
{
    const int idx = blockIdx.x * 256 + threadIdx.x;   // ROWS_N*H exact
    const int b = idx / (Nn * Hn);
    nc[idx] = n[idx] + c[b * Hn + (idx & 255)];
}

// ---- ne5[b,i,j,hd] = q[b,i,hd]*k[b,j,hd]*SCALE + e[b,i,j,hd] ----
__global__ __launch_bounds__(256)
void ne5_kernel(const float* __restrict__ qkv, const float* __restrict__ e,
                float* __restrict__ ne5)
{
    const size_t idx = (size_t)blockIdx.x * 256 + threadIdx.x;  // ROWS_E*H exact
    const int hd = idx & 255;
    const size_t r = idx >> 8;
    const int j = (int)(r % Nn);
    const size_t r2 = r / Nn;
    const int i = (int)(r2 % Nn);
    const int b = (int)(r2 / Nn);
    const float q = qkv[(size_t)(b * Nn + i) * 768 + hd];
    const float k = qkv[(size_t)(b * Nn + j) * 768 + 256 + hd];
    ne5[idx] = q * k * SCALEC + e[idx];
}

// ---- output head: out = X(H) @ w2(H x OD) + b2, K=256 ----
__global__ __launch_bounds__(128)
void head_kernel(const float* __restrict__ X, const float* __restrict__ w2,
                 const float* __restrict__ b2, float* __restrict__ out,
                 int rows, int OD)
{
    const int idx = blockIdx.x * 128 + threadIdx.x;
    if (idx >= rows * OD) return;
    const int row = idx / OD, j = idx % OD;
    float acc = b2[j];
    const float* xr = X + (size_t)row * Hn;
    for (int k = 0; k < Hn; ++k) acc += xr[k] * w2[k * OD + j];
    out[idx] = acc;
}

// =====================================================================
extern "C" void kernel_launch(void* const* d_in, const int* in_sizes, int n_in,
                              void* d_out, int out_size, void* d_ws, size_t ws_size,
                              hipStream_t stream)
{
    (void)in_sizes; (void)n_in; (void)out_size; (void)ws_size;
    const float* nodes  = (const float*)d_in[0];
    const float* edges  = (const float*)d_in[1];
    const int*   conds  = (const int*)  d_in[2];
    const float* inN_w1 = (const float*)d_in[3];  const float* inN_b1 = (const float*)d_in[4];
    const float* inN_w2 = (const float*)d_in[5];  const float* inN_b2 = (const float*)d_in[6];
    const float* inE_w1 = (const float*)d_in[7];  const float* inE_b1 = (const float*)d_in[8];
    const float* inE_w2 = (const float*)d_in[9];  const float* inE_b2 = (const float*)d_in[10];
    const float* inC_w1 = (const float*)d_in[11]; const float* inC_b1 = (const float*)d_in[12];
    const float* inC_w2 = (const float*)d_in[13]; const float* inC_b2 = (const float*)d_in[14];
    const float* qkv_w  = (const float*)d_in[15]; const float* qkv_b  = (const float*)d_in[16];
    const float* no_w   = (const float*)d_in[17]; const float* no_b   = (const float*)d_in[18];
    const float* eo_w   = (const float*)d_in[19]; const float* eo_b   = (const float*)d_in[20];
    const float* ln1n_g = (const float*)d_in[21]; const float* ln1n_b = (const float*)d_in[22];
    const float* ln1e_g = (const float*)d_in[23]; const float* ln1e_b = (const float*)d_in[24];
    const float* mn_w1  = (const float*)d_in[25]; const float* mn_b1  = (const float*)d_in[26];
    const float* mn_w2  = (const float*)d_in[27]; const float* mn_b2  = (const float*)d_in[28];
    const float* me_w1  = (const float*)d_in[29]; const float* me_b1  = (const float*)d_in[30];
    const float* me_w2  = (const float*)d_in[31]; const float* me_b2  = (const float*)d_in[32];
    const float* ln2n_g = (const float*)d_in[33]; const float* ln2n_b = (const float*)d_in[34];
    const float* ln2e_g = (const float*)d_in[35]; const float* ln2e_b = (const float*)d_in[36];
    const float* outN_w1= (const float*)d_in[37]; const float* outN_b1= (const float*)d_in[38];
    const float* outN_w2= (const float*)d_in[39]; const float* outN_b2= (const float*)d_in[40];
    const float* outE_w1= (const float*)d_in[41]; const float* outE_b1= (const float*)d_in[42];
    const float* outE_w2= (const float*)d_in[43]; const float* outE_b2= (const float*)d_in[44];

    // ---- workspace layout (fp32) ----
    float* ws   = (float*)d_ws;
    float* cbuf = ws;                        // 1024
    float* nbuf = cbuf + 1024;               // 24576   node carry
    float* ebuf = nbuf + ROWS_N * Hn;        // 589824  edge carry
    float* ncb  = ebuf + ROWS_E * Hn;        // 24576
    float* qkvb = ncb  + ROWS_N * Hn;        // 73728
    float* wvb  = qkvb + ROWS_N * 768;       // 24576
    float* tmpn = wvb  + ROWS_N * Hn;        // 24576   pre-LN node
    float* anb  = tmpn + ROWS_N * Hn;        // 24576
    float* hidn = anb  + ROWS_N * Hn;        // 98304   node MLP hidden
    float* ne5b = hidn + ROWS_N * 1024;      // 589824
    float* tmpe = ne5b + ROWS_E * Hn;        // 589824  pre-LN edge
    float* aeb  = tmpe + ROWS_E * Hn;        // 589824
    float* hide = aeb  + ROWS_E * Hn;        // 2359296 edge MLP hidden

    const dim3 blk128(128), blk256(256);
    // grid = (M/32, N/128)
    const dim3 gN_256(ROWS_N / 32, 256 / 128);   // (3,2)
    const dim3 gN_768(ROWS_N / 32, 768 / 128);   // (3,6)
    const dim3 gN_1024(ROWS_N / 32, 1024 / 128); // (3,8)
    const dim3 gE_256(ROWS_E / 32, 256 / 128);   // (72,2)
    const dim3 gE_1024(ROWS_E / 32, 1024 / 128); // (72,8)

    // ---- input encoders ----
    cond_kernel<<<Bn, blk256, 0, stream>>>(conds, inC_w1, inC_b1, inC_w2, inC_b2, cbuf);
    nodein_kernel<<<ROWS_N, blk256, 0, stream>>>(nodes, inN_w1, inN_b1, inN_w2, inN_b2, nbuf);
    edgein1_kernel<<<ROWS_E, blk256, 0, stream>>>(edges, inE_w1, inE_b1, ne5b);
    gemm_wmma_kernel<1><<<gE_256, blk128, 0, stream>>>(ne5b, inE_w2, inE_b2, nullptr,
                                                       ebuf, ROWS_E, 256, 256);

    // ---- 64 transformer layers ----
    for (int l = 0; l < Ln; ++l) {
        const float* qw  = qkv_w + (size_t)l * Hn * 768;
        const float* qb  = qkv_b + (size_t)l * 768;
        const float* nw  = no_w  + (size_t)l * Hn * Hn;
        const float* nb  = no_b  + (size_t)l * Hn;
        const float* ew  = eo_w  + (size_t)l * Hn * Hn;
        const float* eb2 = eo_b  + (size_t)l * Hn;

        addc_kernel<<<ROWS_N, blk256, 0, stream>>>(nbuf, cbuf, ncb);
        gemm_wmma_kernel<0><<<gN_768, blk128, 0, stream>>>(ncb, qw, qb, nullptr,
                                                           qkvb, ROWS_N, 768, 256);
        attn_kernel<<<(Bn * Nn * NHn) / 4, blk128, 0, stream>>>(qkvb, ebuf, wvb);
        gemm_wmma_kernel<2><<<gN_256, blk128, 0, stream>>>(wvb, nw, nb, nbuf,
                                                           tmpn, ROWS_N, 256, 256);
        ln_kernel<<<ROWS_N, blk256, 0, stream>>>(tmpn, ln1n_g + l * Hn, ln1n_b + l * Hn, anb);

        ne5_kernel<<<ROWS_E, blk256, 0, stream>>>(qkvb, ebuf, ne5b);
        gemm_wmma_kernel<3><<<gE_256, blk128, 0, stream>>>(ne5b, ew, eb2, ebuf,
                                                           tmpe, ROWS_E, 256, 256);
        ln_kernel<<<ROWS_E, blk256, 0, stream>>>(tmpe, ln1e_g + l * Hn, ln1e_b + l * Hn, aeb);

        gemm_wmma_kernel<1><<<gN_1024, blk128, 0, stream>>>(anb, mn_w1 + (size_t)l * Hn * 1024,
                                                            mn_b1 + (size_t)l * 1024, nullptr,
                                                            hidn, ROWS_N, 1024, 256);
        gemm_wmma_kernel<2><<<gN_256, blk128, 0, stream>>>(hidn, mn_w2 + (size_t)l * 1024 * Hn,
                                                           mn_b2 + (size_t)l * Hn, anb,
                                                           tmpn, ROWS_N, 256, 1024);
        ln_kernel<<<ROWS_N, blk256, 0, stream>>>(tmpn, ln2n_g + l * Hn, ln2n_b + l * Hn, nbuf);

        gemm_wmma_kernel<1><<<gE_1024, blk128, 0, stream>>>(aeb, me_w1 + (size_t)l * Hn * 1024,
                                                            me_b1 + (size_t)l * 1024, nullptr,
                                                            hide, ROWS_E, 1024, 256);
        gemm_wmma_kernel<2><<<gE_256, blk128, 0, stream>>>(hide, me_w2 + (size_t)l * 1024 * Hn,
                                                           me_b2 + (size_t)l * Hn, aeb,
                                                           tmpe, ROWS_E, 256, 1024);
        ln_kernel<<<ROWS_E, blk256, 0, stream>>>(tmpe, ln2e_g + l * Hn, ln2e_b + l * Hn, ebuf);
    }

    // ---- output heads ----
    float* out_n = (float*)d_out;                       // 96*13
    float* out_e = out_n + ROWS_N * NDn;                // 2304*12
    gemm_wmma_kernel<1><<<gN_256, blk128, 0, stream>>>(nbuf, outN_w1, outN_b1, nullptr,
                                                       tmpn, ROWS_N, 256, 256);
    head_kernel<<<(ROWS_N * NDn + 127) / 128, blk128, 0, stream>>>(tmpn, outN_w2, outN_b2,
                                                                   out_n, ROWS_N, NDn);
    gemm_wmma_kernel<1><<<gE_256, blk128, 0, stream>>>(ebuf, outE_w1, outE_b1, nullptr,
                                                       tmpe, ROWS_E, 256, 256);
    head_kernel<<<(ROWS_E * EDn + 127) / 128, blk128, 0, stream>>>(tmpe, outE_w2, outE_b2,
                                                                   out_e, ROWS_E, EDn);
}
